// MultiAttention_19078244728961
// MI455X (gfx1250) — matrix operand order repaired and checked by reading, not compiled, not run on hardware
//
#include <hip/hip_runtime.h>

#define B_  4
#define S_  2048
#define D_  512
#define H_  8
#define HD_ 64

typedef __attribute__((ext_vector_type(16))) _Float16 v16h;
typedef __attribute__((ext_vector_type(8)))  _Float16 h8;
typedef __attribute__((ext_vector_type(8)))  float    v8f;
typedef unsigned int u32x4 __attribute__((ext_vector_type(4)));
typedef int          i32x4 __attribute__((ext_vector_type(4)));
typedef int          i32x8 __attribute__((ext_vector_type(8)));

__device__ __forceinline__ v8f wmma_f16(v16h a, v16h b, v8f c) {
  // D = A(16x32 f16) * B(32x16 f16) + C(16x16 f32)
  return __builtin_amdgcn_wmma_f32_16x16x32_f16(
      /*neg_a=*/false, a, /*neg_b=*/false, b,
      /*c_mod=*/(short)0, c, /*reuse_a=*/false, /*reuse_b=*/false);
}

// A-fragment (16x32 f16) from row-major [M x K] f16, src pre-offset to (row0=0, k0).
// Lane L: M = L%16; elems 0..7: K = 8*(L/16)+i ; elems 8..15: K = 16 + 8*(L/16)+(i-8)
__device__ __forceinline__ v16h load_a_frag(const _Float16* src, int ld) {
  const int lane = threadIdx.x & 31;
  const int m = lane & 15, hi = lane >> 4;
  const _Float16* p = src + m * ld + (hi << 3);
  h8 lo  = *(const h8*)(p);
  h8 hiv = *(const h8*)(p + 16);
  v16h r;
#pragma unroll
  for (int i = 0; i < 8; ++i) { r[i] = lo[i]; r[i + 8] = hiv[i]; }
  return r;
}

// B-fragment (32x16) where memory holds B^T row-major: src[n][k], src pre-offset to (k0, n0).
// Lane L: N = L%16; elems 0..7: K = 8*(L/16)+i ; elems 8..15: K = 16 + 8*(L/16)+(i-8)
__device__ __forceinline__ v16h load_bt_frag(const _Float16* src, int ld) {
  const int lane = threadIdx.x & 31;
  const int n = lane & 15, hi = lane >> 4;
  const _Float16* p = src + n * ld + (hi << 3);
  h8 lo  = *(const h8*)(p);
  h8 hiv = *(const h8*)(p + 16);
  v16h r;
#pragma unroll
  for (int i = 0; i < 8; ++i) { r[i] = lo[i]; r[i + 8] = hiv[i]; }
  return r;
}

// ---- Tensor Data Mover: 2D tile (u64 elements) global -> LDS -------------
// D# per CDNA5 ISA: group0 = {flags, lds_addr, gaddr, type},
// group1 = {data_size, tensor dims, tile dims, dim0 stride}. 2D => groups 2/3 zero.
// This toolchain exposes the 6-arg builtin (extra int32x8 group block, zero-filled).
__device__ __forceinline__ void tdm_load_2d(unsigned lds_addr, unsigned long long gaddr,
                                            unsigned tile_d0, unsigned tile_d1,
                                            unsigned long long stride0,
                                            unsigned tensor_d0, unsigned tensor_d1) {
  u32x4 g0;
  g0[0] = 1u;                                            // count=1, user mode, no gather
  g0[1] = lds_addr;                                      // LDS byte address
  g0[2] = (unsigned)(gaddr & 0xffffffffull);             // global addr [95:64]
  g0[3] = (unsigned)((gaddr >> 32) & 0x01ffffffull)      // global addr [120:96]
        | (2u << 30);                                    // type = 2 ("image")
  i32x8 g1;
  g1[0] = (int)(3u << 16);                               // data_size = 3 (8 bytes)
  g1[1] = (int)((tensor_d0 & 0xffffu) << 16);            // [63:48] tensor_dim0 lo
  g1[2] = (int)((tensor_d0 >> 16) | ((tensor_d1 & 0xffffu) << 16));
  g1[3] = (int)((tensor_d1 >> 16) | (tile_d0 << 16));    // [127:112] tile_dim0
  g1[4] = (int)(tile_d1);                                // tile_dim1 (tile_dim2 = 0)
  g1[5] = (int)(stride0 & 0xffffffffull);                // tensor_dim0_stride lo
  g1[6] = (int)((stride0 >> 32) & 0xffffull);            // stride hi (stride1 = 0)
  g1[7] = 0;
  i32x4 z4 = {0, 0, 0, 0};
  i32x8 z8 = {0, 0, 0, 0, 0, 0, 0, 0};
  __builtin_amdgcn_tensor_load_to_lds(g0, g1, z4, z4, z8, 0);
}

// ---------------- W pre-transpose + f16 convert ----------------
// Wt[p][n][k] f16  <-  W[p][k][n] f32 ;  grid (D*D/256, 3), block 256
__global__ void convert_w_kernel(const float* __restrict__ Wk,
                                 const float* __restrict__ Wv,
                                 const float* __restrict__ Wq,
                                 _Float16* __restrict__ Wt) {
  const int p = blockIdx.y;
  const float* W = (p == 0) ? Wk : (p == 1) ? Wv : Wq;
  _Float16* dst = Wt + (size_t)p * D_ * D_;
  const int idx = blockIdx.x * blockDim.x + threadIdx.x;  // over D*D
  const int k = idx >> 9, n = idx & (D_ - 1);
  dst[(size_t)n * D_ + k] = (_Float16)W[idx];
}

// ---------------- QKV projection: out = X @ W + b, stored f16 ----------------
// grid = (B*S/16, 3), block = 256 (8 waves). Wave w == head w (64 output cols).
// p=0: key->Kh [b][h][s][hd] ; p=1: value->Vt [b][h][hd][s] ; p=2: query->Qh
__global__ void qkv_proj_kernel(const float* __restrict__ key,
                                const float* __restrict__ value,
                                const float* __restrict__ query,
                                const _Float16* __restrict__ Wt,
                                const float* __restrict__ bk,
                                const float* __restrict__ bv,
                                const float* __restrict__ bq,
                                _Float16* __restrict__ Qh,
                                _Float16* __restrict__ Kh,
                                _Float16* __restrict__ Vt) {
  __shared__ __align__(16) _Float16 aTile[16 * 520];  // 16 x 512 (+8 pad, keeps 16B align)

  const int p = blockIdx.y;
  const float* X    = (p == 0) ? key : (p == 1) ? value : query;
  const float* bias = (p == 0) ? bk  : (p == 1) ? bv    : bq;
  const _Float16* Wtp = Wt + (size_t)p * D_ * D_;

  const int r0 = blockIdx.x << 4;  // first global row (b*S + s)
  for (int idx = threadIdx.x; idx < 16 * D_; idx += blockDim.x) {
    const int row = idx >> 9, col = idx & (D_ - 1);
    aTile[row * 520 + col] = (_Float16)X[(size_t)(r0 + row) * D_ + col];
  }
  __syncthreads();

  const int wave = threadIdx.x >> 5;
  const int lane = threadIdx.x & 31;
  const int hi = lane >> 4, n16 = lane & 15;
  const int h = wave;
  const int col0 = h * HD_;

  v8f acc[4] = {};
  for (int k = 0; k < D_; k += 32) {
    v16h a = load_a_frag(&aTile[k], 520);
#pragma unroll
    for (int nt = 0; nt < 4; ++nt) {
      v16h bf = load_bt_frag(Wtp + (size_t)(col0 + nt * 16) * D_ + k, D_);
      acc[nt] = wmma_f16(a, bf, acc[nt]);
    }
  }

  const int b  = r0 >> 11;
  const int s0 = r0 & (S_ - 1);
#pragma unroll
  for (int nt = 0; nt < 4; ++nt) {
    const int d = nt * 16 + n16;
    const float bia = bias[col0 + d];
#pragma unroll
    for (int r = 0; r < 8; ++r) {
      const int s = s0 + r + (hi << 3);
      const _Float16 hv = (_Float16)(acc[nt][r] + bia);
      if (p == 1) {
        Vt[((size_t)(b * H_ + h) * HD_ + d) * S_ + s] = hv;
      } else {
        _Float16* dst = (p == 0) ? Kh : Qh;
        dst[((size_t)(b * H_ + h) * S_ + s) * HD_ + d] = hv;
      }
    }
  }
}

// ---------------- causal flash attention ----------------
// Block = one (b,h) x 64 queries (4 waves x 16-query tile). K/V 32-key blocks are
// TDM-DMA'd into double-buffered LDS and shared by all 4 waves.
// grid = B*H*(S/64) = 1024 blocks, block = 128.
__global__ void flash_attn_kernel(const _Float16* __restrict__ Qh,
                                  const _Float16* __restrict__ Kh,
                                  const _Float16* __restrict__ Vt,
                                  float* __restrict__ out) {
  __shared__ __align__(16) _Float16 kbuf[2][32 * 64];  // 2 x 4KB, ld=64
  __shared__ __align__(16) _Float16 vbuf[2][64 * 32];  // 2 x 4KB, ld=32
  __shared__ __align__(16) _Float16 pl[4][16 * 32];    // per-wave P tile

  const int wave = threadIdx.x >> 5;
  const int lane = threadIdx.x & 31;
  const int hi = lane >> 4, n16 = lane & 15;

  const int bh = blockIdx.x >> 5;          // 32 query-groups per (b,h)
  const int q0 = (blockIdx.x & 31) << 6;   // 64 queries per block
  const int qb = q0 + (wave << 4);         // this wave's 16-query tile

  const _Float16* Q = Qh + (size_t)bh * S_ * HD_;
  const _Float16* K = Kh + (size_t)bh * S_ * HD_;
  const _Float16* V = Vt + (size_t)bh * HD_ * S_;

  const v16h qa0 = load_a_frag(Q + qb * HD_, HD_);       // dims 0..31
  const v16h qa1 = load_a_frag(Q + qb * HD_ + 32, HD_);  // dims 32..63

  float m[8], l[8];
  v8f o[4] = {};
#pragma unroll
  for (int r = 0; r < 8; ++r) { m[r] = -3.0e38f; l[r] = 0.0f; }

  const int kend = ((((q0 + 63) >> 5) + 1) << 5);  // causal bound for whole block

  // prologue: DMA first K/V block into buffer 0 (wave 0 issues; EXEC-independent)
  if (wave == 0) {
    // K block: 32 rows x 64 halves, contiguous -> 2D tile of u64: 16 x 32, stride 16
    tdm_load_2d((unsigned)(size_t)&kbuf[0][0], (unsigned long long)(size_t)K,
                16u, 32u, 16ull, 16u, 32u);
    // V block from Vt: 64 rows x 32 halves, row stride 2048 halves (512 u64)
    tdm_load_2d((unsigned)(size_t)&vbuf[0][0], (unsigned long long)(size_t)V,
                8u, 64u, 512ull, 512u, 64u);
  }

  _Float16* pbuf = pl[wave];
  for (int jb = 0; jb < kend; jb += 32) {
    const int cur = (jb >> 5) & 1;
    if (wave == 0) {
      if (jb + 32 < kend) {  // prefetch next block into the other buffer
        tdm_load_2d((unsigned)(size_t)&kbuf[cur ^ 1][0],
                    (unsigned long long)(size_t)(K + (size_t)(jb + 32) * HD_),
                    16u, 32u, 16ull, 16u, 32u);
        tdm_load_2d((unsigned)(size_t)&vbuf[cur ^ 1][0],
                    (unsigned long long)(size_t)(V + (jb + 32)),
                    8u, 64u, 512ull, 512u, 64u);
        __builtin_amdgcn_s_wait_tensorcnt(2);  // retire current block's 2 DMAs (in-order)
      } else {
        __builtin_amdgcn_s_wait_tensorcnt(0);
      }
    }
    __syncthreads();  // current K/V block visible to all waves

    if (jb <= qb + 15) {  // wave-uniform causal skip
      const _Float16* kb_ = kbuf[cur];
      const _Float16* vb_ = vbuf[cur];

      // ---- scores: S = Q(16x64) . K^T(64x32), fp32 accum
      v8f s0 = {}, s1 = {};
      {
        v16h kf0 = load_bt_frag(kb_, HD_);
        v16h kf1 = load_bt_frag(kb_ + 32, HD_);
        s0 = wmma_f16(qa0, kf0, s0);
        s0 = wmma_f16(qa1, kf1, s0);
      }
      {
        v16h kf0 = load_bt_frag(kb_ + 16 * HD_, HD_);
        v16h kf1 = load_bt_frag(kb_ + 16 * HD_ + 32, HD_);
        s1 = wmma_f16(qa0, kf0, s1);
        s1 = wmma_f16(qa1, kf1, s1);
      }

      // ---- scale + causal mask + online softmax (row = r + 8*hi, col = n16)
      const bool needMask = (jb + 31 > qb);
#pragma unroll
      for (int r = 0; r < 8; ++r) {
        const int q = qb + r + (hi << 3);
        float a0 = s0[r] * 0.125f;  // 1/sqrt(64)
        float a1 = s1[r] * 0.125f;
        if (needMask) {
          if (jb + n16 > q)      a0 = -1.0e30f;
          if (jb + 16 + n16 > q) a1 = -1.0e30f;
        }
        float tm = fmaxf(a0, a1);
        tm = fmaxf(tm, __shfl_xor(tm, 1, 32));
        tm = fmaxf(tm, __shfl_xor(tm, 2, 32));
        tm = fmaxf(tm, __shfl_xor(tm, 4, 32));
        tm = fmaxf(tm, __shfl_xor(tm, 8, 32));
        const float mnew = fmaxf(m[r], tm);
        const float corr = __expf(m[r] - mnew);
        const float p0 = __expf(a0 - mnew);
        const float p1 = __expf(a1 - mnew);
        float rs = p0 + p1;
        rs += __shfl_xor(rs, 1, 32);
        rs += __shfl_xor(rs, 2, 32);
        rs += __shfl_xor(rs, 4, 32);
        rs += __shfl_xor(rs, 8, 32);
        l[r] = l[r] * corr + rs;
        m[r] = mnew;
        const int row = r + (hi << 3);
        pbuf[row * 32 + n16]      = (_Float16)p0;  // C-layout -> A-layout via LDS
        pbuf[row * 32 + 16 + n16] = (_Float16)p1;
        o[0][r] *= corr; o[1][r] *= corr; o[2][r] *= corr; o[3][r] *= corr;
      }

      asm volatile("s_wait_dscnt 0" ::: "memory");  // P stores visible in-wave

      // ---- O += P(16x32) . V(32x64)
      const v16h pf = load_a_frag(pbuf, 32);
#pragma unroll
      for (int nt = 0; nt < 4; ++nt) {
        v16h vf = load_bt_frag(vb_ + nt * 16 * 32, 32);
        o[nt] = wmma_f16(pf, vf, o[nt]);
      }
    }
    __syncthreads();  // all waves done with buf[cur] before it is overwritten
  }

  // ---- normalize and write fp32 output [B][S][D], D index = h*64 + d
  const int b = bh >> 3, h = bh & 7;
#pragma unroll
  for (int nt = 0; nt < 4; ++nt) {
    const int d = h * HD_ + nt * 16 + n16;
#pragma unroll
    for (int r = 0; r < 8; ++r) {
      const int q = qb + r + (hi << 3);
      out[(size_t)(b * S_ + q) * D_ + d] = o[nt][r] / l[r];
    }
  }
}

extern "C" void kernel_launch(void* const* d_in, const int* in_sizes, int n_in,
                              void* d_out, int out_size, void* d_ws, size_t ws_size,
                              hipStream_t stream) {
  (void)in_sizes; (void)n_in; (void)out_size; (void)ws_size;
  const float* key   = (const float*)d_in[0];
  const float* value = (const float*)d_in[1];
  const float* query = (const float*)d_in[2];
  // d_in[3] is the causal mask; it is static (triu, k=1) and folded into the kernels.
  const float* Wk = (const float*)d_in[4];
  const float* bk = (const float*)d_in[5];
  const float* Wv = (const float*)d_in[6];
  const float* bv = (const float*)d_in[7];
  const float* Wq = (const float*)d_in[8];
  const float* bq = (const float*)d_in[9];

  const size_t elems = (size_t)B_ * S_ * D_;  // 4M halves per tensor
  _Float16* Qh = (_Float16*)d_ws;             //  8 MB
  _Float16* Kh = Qh + elems;                  //  8 MB
  _Float16* Vt = Kh + elems;                  //  8 MB
  _Float16* Wt = Vt + elems;                  //  1.5 MB (3 x 512 x 512 f16)

  convert_w_kernel<<<dim3((D_ * D_) / 256, 3), 256, 0, stream>>>(Wk, Wv, Wq, Wt);

  qkv_proj_kernel<<<dim3((B_ * S_) / 16, 3), 256, 0, stream>>>(
      key, value, query, Wt, bk, bv, bq, Qh, Kh, Vt);

  flash_attn_kernel<<<dim3(B_ * H_ * (S_ / 64)), 128, 0, stream>>>(
      Qh, Kh, Vt, (float*)d_out);
}
